// DeformableAttention_89876485636824
// MI455X (gfx1250) — matrix operand order repaired and checked by reading, not compiled, not run on hardware
//
#include <hip/hip_runtime.h>
#include <cstdint>
#include <cstddef>

typedef _Float16 h16;
typedef __attribute__((ext_vector_type(16))) _Float16 v16h;
typedef __attribute__((ext_vector_type(8)))  float    v8f;

#define B_     2
#define C_     512
#define H_     64
#define W_     64
#define HW_    4096
#define HEADS_ 8
#define DK_    64
#define N_     1024      /* sampled kv points per batch (32*32) */
#define R_     63
#define SCALE_ (1.0f/4096.0f)

// ---------------------------------------------------------------------------
// WMMA helpers (CDNA5 wave32, V_WMMA_F32_16X16X32_F16)
// A-frag 16x32 f16: lane l: m=l%16, hi=l/16; elem e<8 -> k=hi*8+e ; e>=8 -> k=hi*8+16+(e-8)
// B-frag 32x16 f16: lane l: n=l%16, hf=l/16; elem e<8 -> k=hf*8+e ; e>=8 -> k=hf*8+16+(e-8)
// C/D  16x16 f32 : lane l: n=l%16;           elem e -> m=e+8*(l/16)
// ---------------------------------------------------------------------------
__device__ __forceinline__ v8f wmma_f16(v16h a, v16h b, v8f c) {
  return __builtin_amdgcn_wmma_f32_16x16x32_f16(false, a, false, b, (short)0, c,
                                                false, false);
}

__device__ __forceinline__ v16h load_afrag(const h16* lane_row) {
  int hi = (threadIdx.x >> 4) & 1;
  const h16* p = lane_row + hi * 8;
  v16h r;
#pragma unroll
  for (int e = 0; e < 8; e++) r[e] = p[e];
#pragma unroll
  for (int e = 0; e < 8; e++) r[8 + e] = p[16 + e];
  return r;
}

__device__ __forceinline__ v16h load_bfrag(const h16* lane_row) {
  int hf = (threadIdx.x >> 4) & 1;
  const h16* p = lane_row + hf * 8;
  v16h r;
#pragma unroll
  for (int e = 0; e < 8; e++) r[e] = p[e];
#pragma unroll
  for (int e = 0; e < 8; e++) r[8 + e] = p[16 + e];
  return r;
}

// CDNA5 async global->LDS copy of 16 bytes (ASYNCcnt-tracked).
// lds_off: byte address inside the workgroup LDS allocation (low 32 bits of a
// generic __shared__ pointer); gaddr: 64-bit global address.
__device__ __forceinline__ void async_copy_b128(unsigned lds_off,
                                                unsigned long long gaddr) {
  asm volatile("global_load_async_to_lds_b128 %0, %1, off"
               :: "v"(lds_off), "v"(gaddr) : "memory");
}
__device__ __forceinline__ void async_wait0() {
  asm volatile("s_wait_asynccnt 0x0" ::: "memory");
}

__device__ __forceinline__ int clampi(int v, int lo, int hi) {
  return v < lo ? lo : (v > hi ? hi : v);
}

// ---------------------------------------------------------------------------
// Packing / conversion kernels
// ---------------------------------------------------------------------------
__global__ void pack_x_kernel(const float* __restrict__ x, h16* __restrict__ x_h) {
  size_t i = (size_t)blockIdx.x * blockDim.x + threadIdx.x;   // over B*C*HW
  if (i >= (size_t)B_ * C_ * HW_) return;
  int p = (int)(i & (HW_ - 1));
  int c = (int)((i >> 12) & (C_ - 1));
  int b = (int)(i >> 21);
  x_h[((size_t)b * HW_ + p) * C_ + c] = (h16)x[i];
}

__global__ void cvt_f16_kernel(const float* __restrict__ in, h16* __restrict__ out, int n) {
  int i = blockIdx.x * blockDim.x + threadIdx.x;
  if (i < n) out[i] = (h16)in[i];
}

// off1_w (O=512, I=512, 2, 2) -> [tap][O][I] f16, tap = dy*2+dx
__global__ void pack_off1_kernel(const float* __restrict__ w, h16* __restrict__ out) {
  int gi = blockIdx.x * blockDim.x + threadIdx.x;   // over 512*512*4
  if (gi >= C_ * C_ * 4) return;
  int tap = gi & 3;
  int oi  = gi >> 2;                                 // o*512 + i
  out[(size_t)tap * C_ * C_ + oi] = (h16)w[gi];
}

// ---------------------------------------------------------------------------
// Generic WMMA GEMM: OUT[M,N] = W[M,K] * B[N,K]^T (+bias[m])
// 4 waves / block, each wave a 32x32 tile; the 4 tiles of a block share the
// same n-tile (tilesM is a multiple of 4), so the B tile is async-staged into
// LDS once per block per k-step.  Grids must supply an exact multiple of 4
// tiles (all launches below do).
// mode 0: out_h[n*512 + m]                          (activations [pix][C])
// mode 1: m<512 -> k_h[n*512+m] ; m>=512 -> v_t[b][m-512][nn]  (kv split)
// mode 2: out_f[(n>>12)*512*4096 + m*4096 + (n&4095)] (proj, f32 NCHW)
// ---------------------------------------------------------------------------
__device__ __forceinline__ void store_tile(v8f acc, int mt, int nt,
                                           const float* __restrict__ bias,
                                           h16* out_h, float* out_f, int mode) {
  int lane = threadIdx.x & 31, lm = lane & 15, lh = lane >> 4;
  int n = nt + lm;
#pragma unroll
  for (int e = 0; e < 8; e++) {
    int m = mt + e + 8 * lh;
    float v = acc[e] + (bias ? bias[m] : 0.0f);
    if (mode == 0) {
      out_h[(size_t)n * C_ + m] = (h16)v;
    } else if (mode == 1) {
      if (m < C_) {
        out_h[(size_t)n * C_ + m] = (h16)v;                       // K: [n][C]
      } else {
        int b = n >> 10, nn = n & 1023;
        out_h[(size_t)B_ * N_ * C_ +
              ((size_t)b * C_ + (m - C_)) * N_ + nn] = (h16)v;    // V^T: [b][C][n]
      }
    } else {
      int b = n >> 12, pp = n & 4095;
      out_f[((size_t)b * C_ + m) * HW_ + pp] = v;                 // NCHW f32
    }
  }
}

__global__ void __launch_bounds__(128)
gemm_kernel(const h16* __restrict__ Wm, const h16* __restrict__ Bm,
            const float* __restrict__ bias, int M, int N, int K,
            h16* out_h, float* out_f, int mode) {
  __shared__ h16 sb[32][32];                       // shared B tile (2 KB)
  int tilesM = M >> 5;
  int tile = blockIdx.x * 4 + (threadIdx.x >> 5);
  int tm = tile % tilesM, tn = tile / tilesM;      // tn uniform across block
  int m0 = tm << 5, n0 = tn << 5;
  int lane = threadIdx.x & 31, lm = lane & 15;

  v8f a00 = {}, a01 = {}, a10 = {}, a11 = {};
  const h16* Wr0 = Wm + (size_t)(m0 + lm) * K;
  const h16* Wr1 = Wm + (size_t)(m0 + 16 + lm) * K;

  int r  = (int)(threadIdx.x >> 2);                // 0..31 : B row within tile
  int sg = (int)(threadIdx.x & 3);                 // 0..3  : 16B segment
  unsigned sbdst = (unsigned)(uintptr_t)&sb[r][sg * 8];
  const h16* bsrc = Bm + (size_t)(n0 + r) * K + sg * 8;

  for (int k0 = 0; k0 < K; k0 += 32) {
    // ---- async-stage shared B tile (32 x 32 f16) into LDS ----
    async_copy_b128(sbdst, (unsigned long long)(uintptr_t)(bsrc + k0));
    if (k0 + 32 < K) {                              // prefetch next A rows
      __builtin_prefetch((const void*)(Wr0 + k0 + 32), 0, 3);
      __builtin_prefetch((const void*)(Wr1 + k0 + 32), 0, 3);
    }
    async_wait0();
    __syncthreads();

    v16h wa0 = load_afrag(Wr0 + k0);
    v16h wa1 = load_afrag(Wr1 + k0);
    v16h bb0 = load_bfrag(&sb[lm][0]);
    v16h bb1 = load_bfrag(&sb[16 + lm][0]);
    a00 = wmma_f16(wa0, bb0, a00);
    a01 = wmma_f16(wa0, bb1, a01);
    a10 = wmma_f16(wa1, bb0, a10);
    a11 = wmma_f16(wa1, bb1, a11);
    __syncthreads();                               // protect sb before rewrite
  }
  store_tile(a00, m0,      n0,      bias, out_h, out_f, mode);
  store_tile(a01, m0,      n0 + 16, bias, out_h, out_f, mode);
  store_tile(a10, m0 + 16, n0,      bias, out_h, out_f, mode);
  store_tile(a11, m0 + 16, n0 + 16, bias, out_h, out_f, mode);
}

// ---------------------------------------------------------------------------
// off1: 2x2 stride-2 conv as 4-tap im2col GEMM. M=512, N=2048 (b*1024+p2), K=2048.
// Output t[b][cout][p2] f32 (+bias).  (B rows are per-lane strided gathers, so
// this one keeps direct VGPR loads.)
// ---------------------------------------------------------------------------
__global__ void __launch_bounds__(128)
off1_gemm_kernel(const h16* __restrict__ Wt, const h16* __restrict__ q_h,
                 const float* __restrict__ bias, float* __restrict__ t) {
  const int tilesM = 16;                 // 512/32
  int tile = blockIdx.x * 4 + (threadIdx.x >> 5);
  int tm = tile % tilesM, tn = tile / tilesM;
  int m0 = tm << 5, n0 = tn << 5;
  int lane = threadIdx.x & 31, lm = lane & 15, lh = lane >> 4;

  int n_a = n0 + lm,      n_b = n0 + 16 + lm;
  int ba = n_a >> 10, pa = n_a & 1023, ya = pa >> 5, xa = pa & 31;
  int bb = n_b >> 10, pb = n_b & 1023, yb = pb >> 5, xb = pb & 31;

  v8f a00 = {}, a01 = {}, a10 = {}, a11 = {};
  for (int k0 = 0; k0 < 2048; k0 += 32) {
    int tap = k0 >> 9, cin0 = k0 & 511;
    int dy = tap >> 1, dx = tap & 1;
    const h16* Wr = Wt + (size_t)tap * C_ * C_ + cin0;
    v16h wa0 = load_afrag(Wr + (size_t)(m0 + lm) * C_);
    v16h wa1 = load_afrag(Wr + (size_t)(m0 + 16 + lm) * C_);
    size_t ra = ((size_t)ba * HW_ + (2 * ya + dy) * W_ + (2 * xa + dx)) * C_ + cin0;
    size_t rb = ((size_t)bb * HW_ + (2 * yb + dy) * W_ + (2 * xb + dx)) * C_ + cin0;
    v16h bb0 = load_bfrag(q_h + ra);
    v16h bb1 = load_bfrag(q_h + rb);
    a00 = wmma_f16(wa0, bb0, a00);
    a01 = wmma_f16(wa0, bb1, a01);
    a10 = wmma_f16(wa1, bb0, a10);
    a11 = wmma_f16(wa1, bb1, a11);
  }
  v8f accs[4] = {a00, a01, a10, a11};
  int  mts[4] = {m0, m0, m0 + 16, m0 + 16};
  int  nts[4] = {n0, n0 + 16, n0, n0 + 16};
#pragma unroll
  for (int f = 0; f < 4; f++) {
#pragma unroll
    for (int e = 0; e < 8; e++) {
      int m = mts[f] + e + 8 * lh;
      int n = nts[f] + lm;
      t[((size_t)(n >> 10) * C_ + m) * N_ + (n & 1023)] = accs[f][e] + bias[m];
    }
  }
}

// ---------------------------------------------------------------------------
// GroupNorm (32 groups) + exact GELU. One block per (b, group).
// ---------------------------------------------------------------------------
__global__ void gn_gelu_kernel(const float* __restrict__ t, const float* __restrict__ gn_w,
                               const float* __restrict__ gn_b, float* __restrict__ t_act) {
  int b = blockIdx.x >> 5, g = blockIdx.x & 31;
  const float* base = t + ((size_t)b * C_ + g * 16) * N_;
  float s = 0.f, s2 = 0.f;
  for (int i = threadIdx.x; i < 16 * N_; i += blockDim.x) {
    float v = base[i]; s += v; s2 += v * v;
  }
  __shared__ float rs[256], rq[256];
  rs[threadIdx.x] = s; rq[threadIdx.x] = s2;
  __syncthreads();
  for (int st = 128; st > 0; st >>= 1) {
    if ((int)threadIdx.x < st) { rs[threadIdx.x] += rs[threadIdx.x + st];
                                 rq[threadIdx.x] += rq[threadIdx.x + st]; }
    __syncthreads();
  }
  float inv_n = 1.0f / (16.0f * N_);
  float mu   = rs[0] * inv_n;
  float var  = rq[0] * inv_n - mu * mu;
  float rstd = rsqrtf(var + 1e-5f);
  float* outb = t_act + ((size_t)b * C_ + g * 16) * N_;
  for (int i = threadIdx.x; i < 16 * N_; i += blockDim.x) {
    int c = g * 16 + (i >> 10);
    float v = (base[i] - mu) * rstd * gn_w[c] + gn_b[c];
    outb[i] = 0.5f * v * (1.0f + erff(v * 0.7071067811865475f));
  }
}

// ---------------------------------------------------------------------------
// off2 (2-channel 1x1 conv, no bias) + reference grid + clip -> pos[b][n][{y,x}]
// One block per (b, p2).
// ---------------------------------------------------------------------------
__global__ void off2_pos_kernel(const float* __restrict__ t_act,
                                const float* __restrict__ w2, float* __restrict__ pos) {
  int b = blockIdx.x >> 10, p2 = blockIdx.x & 1023;
  float s0 = 0.f, s1 = 0.f;
  for (int c = threadIdx.x; c < C_; c += blockDim.x) {
    float v = t_act[((size_t)b * C_ + c) * N_ + p2];
    s0 += v * w2[c];
    s1 += v * w2[C_ + c];
  }
  __shared__ float r0[256], r1[256];
  r0[threadIdx.x] = s0; r1[threadIdx.x] = s1;
  __syncthreads();
  for (int st = 128; st > 0; st >>= 1) {
    if ((int)threadIdx.x < st) { r0[threadIdx.x] += r0[threadIdx.x + st];
                                 r1[threadIdx.x] += r1[threadIdx.x + st]; }
    __syncthreads();
  }
  if (threadIdx.x == 0) {
    int y = p2 >> 5, x = p2 & 31;
    float refy = (y + 0.5f) * (2.0f / 31.0f) - 1.0f;
    float refx = (x + 0.5f) * (2.0f / 31.0f) - 1.0f;
    float py = fminf(fmaxf(r0[0] + refy, -1.0f), 1.0f);
    float px = fminf(fmaxf(r1[0] + refx, -1.0f), 1.0f);
    pos[((size_t)b * N_ + p2) * 2 + 0] = py;
    pos[((size_t)b * N_ + p2) * 2 + 1] = px;
  }
}

// ---------------------------------------------------------------------------
// Bilinear grid-sample of x at pos -> sampled f16 [b*1024+n][C]
// ---------------------------------------------------------------------------
__global__ void sample_kernel(const float* __restrict__ x, const float* __restrict__ pos,
                              h16* __restrict__ samp) {
  int b = blockIdx.x >> 10, n = blockIdx.x & 1023;
  float py = pos[((size_t)b * N_ + n) * 2 + 0];
  float px = pos[((size_t)b * N_ + n) * 2 + 1];
  float xf = (px + 1.0f) * 0.5f * 63.0f;
  float yf = (py + 1.0f) * 0.5f * 63.0f;
  float x0 = floorf(xf), y0 = floorf(yf);
  float wx = xf - x0,    wy = yf - y0;
  int x0i = clampi((int)x0, 0, 63), y0i = clampi((int)y0, 0, 63);
  int x1i = x0i + 1 > 63 ? 63 : x0i + 1;
  int y1i = y0i + 1 > 63 ? 63 : y0i + 1;
  float w00 = (1.f - wx) * (1.f - wy), w01 = wx * (1.f - wy);
  float w10 = (1.f - wx) * wy,         w11 = wx * wy;
  const float* xb = x + (size_t)b * C_ * HW_;
  for (int c = threadIdx.x; c < C_; c += blockDim.x) {
    const float* xc = xb + (size_t)c * HW_;
    float v = xc[y0i * 64 + x0i] * w00 + xc[y0i * 64 + x1i] * w01 +
              xc[y1i * 64 + x0i] * w10 + xc[y1i * 64 + x1i] * w11;
    samp[((size_t)b * N_ + n) * C_ + c] = (h16)v;
  }
}

// ---------------------------------------------------------------------------
// Flash attention with deformable rpe bias.
// grid: 16 (b*head) x 64 m-blocks; block = 128 (4 waves), wave = one 16-row m-tile.
// All 4 waves share (b, head) and the n-tile sequence, so K/V tiles are
// async-staged into LDS once per block per 32-wide n-step.
// ---------------------------------------------------------------------------
__device__ __forceinline__ float rpe_bilin(const float* __restrict__ rp,
                                           float gx, float gy) {
  float xf = (gx + 1.0f) * 31.0f;     // 0.5*(R-1), R=63
  float yf = (gy + 1.0f) * 31.0f;
  float x0 = floorf(xf), y0 = floorf(yf);
  float wx = xf - x0,    wy = yf - y0;
  int x0i = clampi((int)x0, 0, 62), y0i = clampi((int)y0, 0, 62);
  int x1i = x0i + 1 > 62 ? 62 : x0i + 1;
  int y1i = y0i + 1 > 62 ? 62 : y0i + 1;
  float v00 = rp[y0i * R_ + x0i], v01 = rp[y0i * R_ + x1i];
  float v10 = rp[y1i * R_ + x0i], v11 = rp[y1i * R_ + x1i];
  return v00 * (1.f - wx) * (1.f - wy) + v01 * wx * (1.f - wy) +
         v10 * (1.f - wx) * wy + v11 * wx * wy;
}

__global__ void __launch_bounds__(128)
attn_kernel(const h16* __restrict__ q_h, const h16* __restrict__ k_h,
            const h16* __restrict__ v_t, const float* __restrict__ pos,
            const float* __restrict__ rpe, h16* __restrict__ o_h) {
  __shared__ h16 sk[32][64];          // K tile  [n_local][c]   (4 KB)
  __shared__ h16 sv[64][32];          // V^T tile [c][n_local]  (4 KB)
  __shared__ float sp[4][16][33];     // per-wave P transpose stage
  int bh = blockIdx.x >> 6;
  int b = bh >> 3, head = bh & 7;
  int wv = threadIdx.x >> 5;
  int m0 = ((blockIdx.x & 63) << 6) + (wv << 4);
  int lane = threadIdx.x & 31, lm = lane & 15, lh = lane >> 4;

  const h16* qbase = q_h + ((size_t)b * HW_ + m0 + lm) * C_ + head * DK_;
  v16h qa0 = load_afrag(qbase);
  v16h qa1 = load_afrag(qbase + 32);

  float qgy[8], qgx[8];
#pragma unroll
  for (int e = 0; e < 8; e++) {
    int m = m0 + e + 8 * lh;
    qgy[e] = (float)(m >> 6) * (2.0f / 63.0f) - 1.0f;
    qgx[e] = (float)(m & 63) * (2.0f / 63.0f) - 1.0f;
  }

  float rmax[8], rsum[8];
#pragma unroll
  for (int e = 0; e < 8; e++) { rmax[e] = -3.0e38f; rsum[e] = 0.0f; }
  v8f o0 = {}, o1 = {}, o2 = {}, o3 = {};

  const float* rp = rpe + head * R_ * R_;

  // per-thread staging assignments (2 x 16B chunks each for K and V^T)
  int tid = threadIdx.x;
  const char* kgbase = (const char*)(k_h + ((size_t)b * N_) * C_ + head * DK_);
  const char* vgbase = (const char*)(v_t + ((size_t)b * C_ + head * DK_) * N_);

  for (int n0 = 0; n0 < N_; n0 += 32) {
    // ---- async-stage K (32x64 f16) and V^T (64x32 f16) tiles ----
#pragma unroll
    for (int u = 0; u < 2; u++) {
      int ck = tid + u * 128;                    // 0..255 chunks of K tile
      int kr = ck >> 3, ks = ck & 7;
      async_copy_b128((unsigned)(uintptr_t)&sk[kr][ks * 8],
                      (unsigned long long)(uintptr_t)
                      (kgbase + ((size_t)(n0 + kr) * C_) * 2 + ks * 16));
      int cv = tid + u * 128;                    // 0..255 chunks of V tile
      int vc = cv >> 2, vs = cv & 3;
      async_copy_b128((unsigned)(uintptr_t)&sv[vc][vs * 8],
                      (unsigned long long)(uintptr_t)
                      (vgbase + ((size_t)vc * N_ + n0) * 2 + vs * 16));
    }
    async_wait0();
    __syncthreads();

    // ---- scores: S = Q^T K (two 16-wide n tiles) from LDS ----
    v8f s0 = {}, s1 = {};
    v16h kf;
    kf = load_bfrag(&sk[lm][0]);        s0 = wmma_f16(qa0, kf, s0);
    kf = load_bfrag(&sk[lm][32]);       s0 = wmma_f16(qa1, kf, s0);
    kf = load_bfrag(&sk[16 + lm][0]);   s1 = wmma_f16(qa0, kf, s1);
    kf = load_bfrag(&sk[16 + lm][32]);  s1 = wmma_f16(qa1, kf, s1);

    float py0 = pos[((size_t)b * N_ + n0 + lm) * 2 + 0];
    float px0 = pos[((size_t)b * N_ + n0 + lm) * 2 + 1];
    float py1 = pos[((size_t)b * N_ + n0 + 16 + lm) * 2 + 0];
    float px1 = pos[((size_t)b * N_ + n0 + 16 + lm) * 2 + 1];

    // ---- bias + online softmax (rows live in 16-lane halves) ----
    float corr[8];
#pragma unroll
    for (int e = 0; e < 8; e++) {
      float v0 = s0[e] * SCALE_ +
                 rpe_bilin(rp, (qgx[e] - px0) * 0.5f, (qgy[e] - py0) * 0.5f);
      float v1 = s1[e] * SCALE_ +
                 rpe_bilin(rp, (qgx[e] - px1) * 0.5f, (qgy[e] - py1) * 0.5f);
      float mx = fmaxf(v0, v1);
#pragma unroll
      for (int d = 1; d < 16; d <<= 1) mx = fmaxf(mx, __shfl_xor(mx, d, 16));
      float nm = fmaxf(rmax[e], mx);
      corr[e] = __expf(rmax[e] - nm);
      rmax[e] = nm;
      float p0 = __expf(v0 - nm), p1 = __expf(v1 - nm);
      float ts = p0 + p1;
#pragma unroll
      for (int d = 1; d < 16; d <<= 1) ts += __shfl_xor(ts, d, 16);
      rsum[e] = rsum[e] * corr[e] + ts;
      s0[e] = p0; s1[e] = p1;
    }
#pragma unroll
    for (int e = 0; e < 8; e++) {
      o0[e] *= corr[e]; o1[e] *= corr[e]; o2[e] *= corr[e]; o3[e] *= corr[e];
    }

    // ---- redistribute P: D-frag layout -> A-frag layout via per-wave LDS ----
    float (*spw)[33] = sp[wv];
#pragma unroll
    for (int e = 0; e < 8; e++) {
      spw[e + 8 * lh][lm]      = s0[e];
      spw[e + 8 * lh][16 + lm] = s1[e];
    }
    asm volatile("s_wait_dscnt 0x0" ::: "memory");
    v16h pa;
#pragma unroll
    for (int e = 0; e < 8; e++) pa[e]     = (h16)spw[lm][lh * 8 + e];
#pragma unroll
    for (int e = 0; e < 8; e++) pa[8 + e] = (h16)spw[lm][lh * 8 + 16 + e];

    // ---- O += P (16x32) * V^T (32 x 16c), 4 c-tiles from LDS ----
    v16h vf;
    vf = load_bfrag(&sv[lm][0]);       o0 = wmma_f16(pa, vf, o0);
    vf = load_bfrag(&sv[16 + lm][0]);  o1 = wmma_f16(pa, vf, o1);
    vf = load_bfrag(&sv[32 + lm][0]);  o2 = wmma_f16(pa, vf, o2);
    vf = load_bfrag(&sv[48 + lm][0]);  o3 = wmma_f16(pa, vf, o3);
    __syncthreads();                   // protect sk/sv before next staging
  }

  // ---- normalize and store O^T rows: o_h[b*4096+m][head*64+c] ----
#pragma unroll
  for (int e = 0; e < 8; e++) {
    float inv = 1.0f / rsum[e];
    int m = m0 + e + 8 * lh;
    h16* orow = o_h + ((size_t)b * HW_ + m) * C_ + head * DK_;
    orow[lm]      = (h16)(o0[e] * inv);
    orow[16 + lm] = (h16)(o1[e] * inv);
    orow[32 + lm] = (h16)(o2[e] * inv);
    orow[48 + lm] = (h16)(o3[e] * inv);
  }
}

// ---------------------------------------------------------------------------
// Host launcher
// ---------------------------------------------------------------------------
extern "C" void kernel_launch(void* const* d_in, const int* in_sizes, int n_in,
                              void* d_out, int out_size, void* d_ws, size_t ws_size,
                              hipStream_t stream) {
  const float* x      = (const float*)d_in[0];
  const float* q_w    = (const float*)d_in[1];
  const float* q_b    = (const float*)d_in[2];
  const float* kv_w   = (const float*)d_in[3];
  const float* kv_b   = (const float*)d_in[4];
  const float* off1_w = (const float*)d_in[5];
  const float* off1_b = (const float*)d_in[6];
  const float* gn_w   = (const float*)d_in[7];
  const float* gn_b   = (const float*)d_in[8];
  const float* off2_w = (const float*)d_in[9];
  const float* rpe    = (const float*)d_in[10];
  const float* proj_w = (const float*)d_in[11];
  const float* proj_b = (const float*)d_in[12];
  float* out = (float*)d_out;

  char* ws = (char*)d_ws;
  size_t off = 0;
  auto alloc = [&](size_t bytes) -> void* {
    void* p = ws + off;
    off += (bytes + 255) & ~(size_t)255;
    return p;
  };
  h16*   x_h   = (h16*)alloc((size_t)B_ * HW_ * C_ * 2);     // 8 MB
  h16*   q_h   = (h16*)alloc((size_t)B_ * HW_ * C_ * 2);     // 8 MB
  h16*   wq_h  = (h16*)alloc((size_t)C_ * C_ * 2);
  h16*   wkv_h = (h16*)alloc((size_t)2 * C_ * C_ * 2);
  h16*   w1_h  = (h16*)alloc((size_t)4 * C_ * C_ * 2);
  h16*   wp_h  = (h16*)alloc((size_t)C_ * C_ * 2);
  float* t     = (float*)alloc((size_t)B_ * C_ * N_ * 4);
  float* t_act = (float*)alloc((size_t)B_ * C_ * N_ * 4);
  float* posb  = (float*)alloc((size_t)B_ * N_ * 2 * 4);
  h16*   samp  = (h16*)alloc((size_t)B_ * N_ * C_ * 2);
  h16*   k_h   = (h16*)alloc((size_t)B_ * N_ * C_ * 2);      // v_t MUST follow
  h16*   v_t   = (h16*)alloc((size_t)B_ * N_ * C_ * 2);
  h16*   o_h   = (h16*)alloc((size_t)B_ * HW_ * C_ * 2);

  // ---- pack ----
  pack_x_kernel<<<(B_ * C_ * HW_ + 255) / 256, 256, 0, stream>>>(x, x_h);
  cvt_f16_kernel<<<(C_ * C_ + 255) / 256, 256, 0, stream>>>(q_w, wq_h, C_ * C_);
  cvt_f16_kernel<<<(2 * C_ * C_ + 255) / 256, 256, 0, stream>>>(kv_w, wkv_h, 2 * C_ * C_);
  cvt_f16_kernel<<<(C_ * C_ + 255) / 256, 256, 0, stream>>>(proj_w, wp_h, C_ * C_);
  pack_off1_kernel<<<(4 * C_ * C_ + 255) / 256, 256, 0, stream>>>(off1_w, w1_h);

  // ---- q = Wq * x  (M=512, N=8192, K=512) ----
  gemm_kernel<<<(16 * 256) / 4, 128, 0, stream>>>(wq_h, x_h, q_b, 512, 8192, 512,
                                                  q_h, nullptr, 0);
  // ---- off1 conv (GEMM over 4 taps) ----
  off1_gemm_kernel<<<(16 * 64) / 4, 128, 0, stream>>>(w1_h, q_h, off1_b, t);
  // ---- GroupNorm + GELU ----
  gn_gelu_kernel<<<B_ * 32, 256, 0, stream>>>(t, gn_w, gn_b, t_act);
  // ---- off2 + reference grid -> pos ----
  off2_pos_kernel<<<B_ * N_, 256, 0, stream>>>(t_act, off2_w, posb);
  // ---- deformable grid sample ----
  sample_kernel<<<B_ * N_, 128, 0, stream>>>(x, posb, samp);
  // ---- kv = Wkv * sampled (M=1024, N=2048, K=512), split K / V^T ----
  gemm_kernel<<<(32 * 64) / 4, 128, 0, stream>>>(wkv_h, samp, kv_b, 1024, 2048, 512,
                                                 k_h, nullptr, 1);
  // ---- flash attention with rpe bias ----
  attn_kernel<<<16 * 64, 128, 0, stream>>>(q_h, k_h, v_t, posb, rpe, o_h);
  // ---- proj (M=512, N=8192, K=512) -> f32 NCHW output ----
  gemm_kernel<<<(16 * 256) / 4, 128, 0, stream>>>(wp_h, o_h, proj_b, 512, 8192, 512,
                                                  nullptr, out, 2);
}
